// ShardAttention_14800457302368
// MI455X (gfx1250) — compile-verified, matrix operand order checked by reading, not run on hardware
//
#include <hip/hip_runtime.h>
#include <hip/hip_bf16.h>
#include <math.h>
#include <stdint.h>

// ---- problem constants (from reference) ----
#define NH   16
#define EMB  1024
#define HS   64        // EMB / NH
#define BB   2
#define TT   2048
#define KD   (NH*TT)   // 32768  contraction dim of final GEMM
#define KW   (NH*EMB)  // 16384  rows of weight

#define LDS_PITCH 40   // bf16 elements per LDS row (32 data + 8 pad, 16B-aligned)

typedef __attribute__((ext_vector_type(16))) __bf16 v16bf;
typedef __attribute__((ext_vector_type(8)))  __bf16 v8bf;
typedef __attribute__((ext_vector_type(8)))  float  v8f;

__device__ __forceinline__ int lane_id() { return (int)(threadIdx.x & 31u); }

__device__ __forceinline__ v8f wmma_bf16(v16bf a, v16bf b, v8f c) {
  // D = A(16x32 bf16) x B(32x16 bf16) + C(16x16 f32)
  return __builtin_amdgcn_wmma_f32_16x16x32_bf16(false, a, false, b,
                                                 (short)0, c, false, false);
}

// Combine two contiguous 8-element K-chunks (k = hf*8.. and k = 16+hf*8..)
// into the 16-bit fragment register layout.
__device__ __forceinline__ v16bf frag_from_chunks(v8bf lo, v8bf hi) {
  v16bf f;
#pragma unroll
  for (int i = 0; i < 8; ++i) { f[i] = lo[i]; f[8 + i] = hi[i]; }
  return f;
}

// A fragment 16x32 (MxK), row-major, leading dim lda. lane: M = l%16, hf = l/16.
__device__ __forceinline__ v16bf load_a_rm(const __bf16* __restrict__ A, int lda) {
  int l = lane_id(); int m = l & 15; int hf = l >> 4;
  const __bf16* p = A + (size_t)m * lda + hf * 8;
  return frag_from_chunks(*(const v8bf*)p, *(const v8bf*)(p + 16));
}

// B fragment 32x16 (KxN) from N-major memory: element (k,n) at Bm[n*ldk + k].
__device__ __forceinline__ v16bf load_b_tn(const __bf16* __restrict__ Bm, int ldk) {
  int l = lane_id(); int n = l & 15; int hf = l >> 4;
  const __bf16* p = Bm + (size_t)n * ldk + hf * 8;
  return frag_from_chunks(*(const v8bf*)p, *(const v8bf*)(p + 16));
}

// B fragment from LDS panel [128 rows(n-local)][LDS_PITCH].
__device__ __forceinline__ v16bf load_b_lds(const __bf16* Bl, int nrow0) {
  int l = lane_id(); int n = l & 15; int hf = l >> 4;
  const __bf16* p = Bl + (size_t)(nrow0 + n) * LDS_PITCH + hf * 8;
  return frag_from_chunks(*(const v8bf*)p, *(const v8bf*)(p + 16));
}

// ---- CDNA5 async global->LDS staging (ASYNCcnt path, no VGPR data) ----
__device__ __forceinline__ void async_b128_to_lds(uint32_t lds_off, const void* gptr) {
  uint64_t g = (uint64_t)(uintptr_t)gptr;
  asm volatile("global_load_async_to_lds_b128 %0, %1, off"
               :: "v"(lds_off), "v"(g) : "memory");
}
__device__ __forceinline__ void wait_async0() {
  asm volatile("s_wait_asynccnt 0" ::: "memory");
}

// Stage a 32(K) x 128(N) bf16 panel: 512 16-byte chunks across 256 threads.
// gbase points at (n=0, k=0) of the panel in N-major memory with leading dim ldk.
__device__ __forceinline__ void panel_fill_async(uint32_t lds_base,
                                                 const __bf16* gbase, size_t ldk) {
  int tid = threadIdx.x;
#pragma unroll
  for (int c = 0; c < 2; ++c) {
    int idx = tid + c * 256;
    int nl = idx >> 2, kq = idx & 3;
    async_b128_to_lds(lds_base + (uint32_t)(nl * LDS_PITCH + kq * 8) * 2,
                      gbase + (size_t)nl * ldk + kq * 8);
  }
}

// ---------------- kernel 1: f32 -> bf16 conversion ----------------
__global__ void sa_cvt_bf16(const float* __restrict__ in, __bf16* __restrict__ out, int n) {
  int i = blockIdx.x * blockDim.x + threadIdx.x;
  int stride = gridDim.x * blockDim.x;
  for (; i < n; i += stride) out[i] = (__bf16)in[i];
}

// ---------------- kernel 2: transpose weight f32[KW][EMB] -> bf16 wbT[EMB][KW] ----
__global__ __launch_bounds__(256)
void sa_cvt_transpose(const float* __restrict__ in, __bf16* __restrict__ out) {
  __shared__ __bf16 tile[32][33];
  int c0 = blockIdx.x * 32;
  int r0 = blockIdx.y * 32;
#pragma unroll
  for (int i = threadIdx.y; i < 32; i += 8)
    tile[i][threadIdx.x] = (__bf16)in[(size_t)(r0 + i) * EMB + (c0 + threadIdx.x)];
  __syncthreads();
#pragma unroll
  for (int i = threadIdx.y; i < 32; i += 8)
    out[(size_t)(c0 + i) * KW + (r0 + threadIdx.x)] = tile[threadIdx.x][i];
}

// ---------------- kernel 3: QK^T / 8 + softmax -> attn (bf16) ----------------
__global__ __launch_bounds__(32)
void sa_qk_softmax(const __bf16* __restrict__ xb, __bf16* __restrict__ attn) {
  const int tiles_t = TT / 16;
  int tile = blockIdx.x;
  int mt = tile % tiles_t;
  int h  = (tile / tiles_t) % NH;
  int b  = tile / (tiles_t * NH);

  const int l = lane_id();
  const int n = l & 15;
  const int hf = l >> 4;

  const __bf16* Q = xb + ((size_t)b*TT + (size_t)mt*16) * EMB + h*HS;
  v16bf aq0 = load_a_rm(Q, EMB);
  v16bf aq1 = load_a_rm(Q + 32, EMB);

  float rmax[8], rsum[8];
#pragma unroll
  for (int r = 0; r < 8; ++r) { rmax[r] = -INFINITY; rsum[r] = 0.0f; }

  // pass 1: online per-row max / sum of exp
  for (int st = 0; st < tiles_t; ++st) {
    const __bf16* Kb = xb + ((size_t)b*TT + (size_t)st*16) * EMB + h*HS;
    __builtin_prefetch(Kb + 16 * EMB, 0, 1);    // next K tile -> global_prefetch_b8
    v8f c = {};
    c = wmma_bf16(aq0, load_b_tn(Kb, EMB), c);
    c = wmma_bf16(aq1, load_b_tn(Kb + 32, EMB), c);
#pragma unroll
    for (int r = 0; r < 8; ++r) {
      float sc = c[r] * 0.125f;
      float mnew = fmaxf(rmax[r], sc);
      rsum[r] = rsum[r] * __expf(rmax[r] - mnew) + __expf(sc - mnew);
      rmax[r] = mnew;
    }
  }

  // reduce across the 16 lanes sharing each row
#pragma unroll
  for (int off = 1; off < 16; off <<= 1) {
#pragma unroll
    for (int r = 0; r < 8; ++r) {
      float om = __shfl_xor(rmax[r], off);
      float os = __shfl_xor(rsum[r], off);
      float mnew = fmaxf(rmax[r], om);
      rsum[r] = rsum[r] * __expf(rmax[r] - mnew) + os * __expf(om - mnew);
      rmax[r] = mnew;
    }
  }
  float rinv[8];
#pragma unroll
  for (int r = 0; r < 8; ++r) rinv[r] = 1.0f / rsum[r];

  // pass 2: recompute, normalize, store bf16 attn
  for (int st = 0; st < tiles_t; ++st) {
    const __bf16* Kb = xb + ((size_t)b*TT + (size_t)st*16) * EMB + h*HS;
    v8f c = {};
    c = wmma_bf16(aq0, load_b_tn(Kb, EMB), c);
    c = wmma_bf16(aq1, load_b_tn(Kb + 32, EMB), c);
#pragma unroll
    for (int r = 0; r < 8; ++r) {
      int t = mt * 16 + r + hf * 8;
      int s = st * 16 + n;
      float p = __expf(c[r] * 0.125f - rmax[r]) * rinv[r];
      attn[(((size_t)b*TT + t) * NH + h) * TT + s] = (__bf16)p;
    }
  }
}

// ---------------- kernel 4: V'[b,h] = x[b] @ W_h, stored transposed ----------
// WG = 8 waves -> 128x128 block; B panel async-staged into double-buffered LDS.
__global__ __launch_bounds__(256)
void sa_vproj(const __bf16* __restrict__ xb, const __bf16* __restrict__ wbT,
              __bf16* __restrict__ vpT) {
  __shared__ __attribute__((aligned(16))) __bf16 Bl[2][128 * LDS_PITCH];

  int blk = blockIdx.x;
  int nb = blk & 7;
  int mb = (blk >> 3) & 15;
  int h  = (blk >> 7) & 15;
  int b  = blk >> 11;

  int wv = threadIdx.x >> 5;
  int s0 = mb * 128 + wv * 16;
  int e0 = nb * 128;

  const int l = lane_id(); const int n = l & 15; const int hf = l >> 4;

  v8f acc[8];
#pragma unroll
  for (int j = 0; j < 8; ++j) acc[j] = (v8f){};

  const __bf16* Arow   = xb + ((size_t)b*TT + s0) * EMB;
  const __bf16* Bpanel = wbT + (size_t)e0 * KW + (size_t)h * EMB;
  uint32_t lds0 = (uint32_t)(uintptr_t)&Bl[0][0];
  uint32_t lds1 = (uint32_t)(uintptr_t)&Bl[1][0];

  panel_fill_async(lds0, Bpanel, KW);
  int cur = 0;
  for (int k0 = 0; k0 < EMB; k0 += 32) {
    wait_async0();
    __syncthreads();
    if (k0 + 32 < EMB)
      panel_fill_async(cur ? lds0 : lds1, Bpanel + k0 + 32, KW);

    v16bf a = load_a_rm(Arow + k0, EMB);
    const __bf16* Bc = &Bl[cur][0];
#pragma unroll
    for (int j = 0; j < 8; ++j)
      acc[j] = wmma_bf16(a, load_b_lds(Bc, j * 16), acc[j]);
    cur ^= 1;
  }

  // transposed store: accumulator rows are consecutive s -> one b128 store each
#pragma unroll
  for (int j = 0; j < 8; ++j) {
    int e = e0 + j * 16 + n;
    v8bf pk;
#pragma unroll
    for (int r = 0; r < 8; ++r) pk[r] = (__bf16)acc[j][r];
    *(v8bf*)(vpT + ((size_t)b*EMB + e) * KD + (size_t)h*TT + s0 + hf*8) = pk;
  }
}

// ---------------- kernel 5: out[b] = attn[b] @ V'[b]  (f32) ------------------
__global__ __launch_bounds__(256)
void sa_final(const __bf16* __restrict__ attn, const __bf16* __restrict__ vpT,
              float* __restrict__ out) {
  __shared__ __attribute__((aligned(16))) __bf16 Bl[2][128 * LDS_PITCH];

  int blk = blockIdx.x;
  int nb = blk & 7;
  int mb = (blk >> 3) & 15;
  int b  = blk >> 7;

  int wv = threadIdx.x >> 5;
  int t0 = mb * 128 + wv * 16;
  int e0 = nb * 128;

  const int l = lane_id(); const int n = l & 15; const int hf = l >> 4;

  v8f acc[8];
#pragma unroll
  for (int j = 0; j < 8; ++j) acc[j] = (v8f){};

  const __bf16* Arow   = attn + ((size_t)b*TT + t0) * KD;
  const __bf16* Bpanel = vpT + ((size_t)b * EMB + e0) * KD;
  uint32_t lds0 = (uint32_t)(uintptr_t)&Bl[0][0];
  uint32_t lds1 = (uint32_t)(uintptr_t)&Bl[1][0];

  panel_fill_async(lds0, Bpanel, KD);
  int cur = 0;
  for (int k0 = 0; k0 < KD; k0 += 32) {
    wait_async0();
    __syncthreads();
    if (k0 + 32 < KD)
      panel_fill_async(cur ? lds0 : lds1, Bpanel + k0 + 32, KD);

    v16bf a = load_a_rm(Arow + k0, KD);
    const __bf16* Bc = &Bl[cur][0];
#pragma unroll
    for (int j = 0; j < 8; ++j)
      acc[j] = wmma_bf16(a, load_b_lds(Bc, j * 16), acc[j]);
    cur ^= 1;
  }

#pragma unroll
  for (int j = 0; j < 8; ++j) {
#pragma unroll
    for (int r = 0; r < 8; ++r) {
      int t = t0 + r + hf * 8;
      out[((size_t)b*TT + t) * EMB + e0 + j*16 + n] = acc[j][r];
    }
  }
}

extern "C" void kernel_launch(void* const* d_in, const int* in_sizes, int n_in,
                              void* d_out, int out_size, void* d_ws, size_t ws_size,
                              hipStream_t stream) {
  const float* x   = (const float*)d_in[0];     // [B, T, EMB] f32
  const float* wgt = (const float*)d_in[1];     // [KW, EMB] f32
  float* out = (float*)d_out;                   // [B, T, EMB] f32

  // workspace (bf16): xb | wbT | attn | vpT
  __bf16* xb   = (__bf16*)d_ws;                          //  4 Mi elems
  __bf16* wbT  = xb   + (size_t)BB * TT * EMB;           // 16 Mi elems [EMB][KW]
  __bf16* attn = wbT  + (size_t)EMB * KW;                // 128 Mi elems [b][t][h][s]
  __bf16* vpT  = attn + (size_t)BB * TT * KD;            // 64 Mi elems [b][e][kg]

  sa_cvt_bf16<<<4096, 256, 0, stream>>>(x, xb, BB * TT * EMB);
  {
    dim3 tb(32, 8), tg(EMB / 32, KW / 32);
    sa_cvt_transpose<<<tg, tb, 0, stream>>>(wgt, wbT);
  }
  sa_qk_softmax<<<BB * NH * (TT / 16), 32, 0, stream>>>(xb, attn);
  sa_vproj<<<BB * NH * 16 * 8, 256, 0, stream>>>(xb, wbT, vpT);
  sa_final<<<BB * 16 * 8, 256, 0, stream>>>(attn, vpT, out);
}
